// Favor_87016037417543
// MI455X (gfx1250) — matrix operand order repaired and checked by reading, not compiled, not run on hardware
//
#include <hip/hip_runtime.h>
#include <hip/hip_bf16.h>
#include <math.h>

typedef __attribute__((ext_vector_type(16))) __bf16 v16bf;
typedef __attribute__((ext_vector_type(8)))  float  v8f;
typedef __attribute__((ext_vector_type(4)))  int    v4i;

#define NE 1024      // n_embd
#define DH 64        // d_head
#define ROWS_PER_BLK 128
#define NKC 32       // number of 32-wide K chunks
#define CHUNK_BYTES 12288   // 12 col-tiles * 32 lanes * 16 bf16 * 2B

#if __has_builtin(__builtin_amdgcn_global_load_async_to_lds_b128)
#  define HAS_ASYNC_LDS 1
typedef __attribute__((address_space(1))) v4i* gp_v4i;
typedef __attribute__((address_space(3))) v4i* lp_v4i;
#  define ASYNC_CP16(gsrc, ldst) \
      __builtin_amdgcn_global_load_async_to_lds_b128((gp_v4i)(gsrc), (lp_v4i)(ldst), 0, 0)
#else
#  define HAS_ASYNC_LDS 0
#endif

#if __has_builtin(__builtin_amdgcn_s_wait_asynccnt)
#  define WAIT_ASYNC(n) __builtin_amdgcn_s_wait_asynccnt(n)
#else
#  define WAIT_ASYNC(n) asm volatile("s_wait_asynccnt %0" :: "n"(n) : "memory")
#endif

// ---------------------------------------------------------------------------
// Kernel 0 (prep, 384 KB once): swizzle [Wq|Wk|Wv] (f32 [1024][64] each) into
// bf16 WMMA B-fragment layout:
//   wswz[kc][ct][lane][e] = W_{ct/4}[kc*32 + 16*(lane>>4) + e][(ct%4)*16 + (lane&15)]
// One thread builds one 16-element fragment (one v16bf store).
// ---------------------------------------------------------------------------
__global__ __launch_bounds__(256) void favor_wswz_kernel(
    const float* __restrict__ Wq, const float* __restrict__ Wk,
    const float* __restrict__ Wv, __bf16* __restrict__ wswz)
{
    int g = blockIdx.x * 256 + threadIdx.x;       // 0 .. 12287 fragments
    int kc = g / 384;                              // 384 = 12*32 fragments/chunk
    int r  = g - kc * 384;
    int ct = r >> 5;
    int ln = r & 31;
    int mtx = ct >> 2;
    int col = ((ct & 3) << 4) + (ln & 15);
    int k0  = kc * 32 + ((ln >> 4) << 4);
    const float* Wm = (mtx == 0) ? Wq : ((mtx == 1) ? Wk : Wv);

    v16bf frag;
    #pragma unroll
    for (int e = 0; e < 16; ++e)
        frag[e] = (__bf16)Wm[(size_t)(k0 + e) * DH + col];
    ((v16bf*)wswz)[g] = frag;
}

// ---------------------------------------------------------------------------
// Kernel 1: fused q/k/v projection (bf16 WMMA, f32 accum) + FAVOR features +
// per-block partial S = Kp^T @ V.
// Block: 256 threads (8 waves). Wave w owns rows [blk*128 + 16w, +16).
// B fragments stream pre-swizzled from L2 via async-to-LDS, double-buffered.
// ---------------------------------------------------------------------------
__global__ __launch_bounds__(256) void favor_proj_kernel(
    const float* __restrict__ X,       // [32768][1024]
    const float* __restrict__ w,       // [64][8]
    const float* __restrict__ bq, const float* __restrict__ bk,
    const float* __restrict__ bv,
    const __bf16* __restrict__ wswz,   // pre-swizzled weights
    float* __restrict__ qp,            // [32768][16]
    float* __restrict__ sPart)         // [gridDim.x][16][64]
{
    // Arena: K-loop uses 2 x 12 KB B-fragment buffers; epilogue reuses it as
    // per-wave 16x64 f32 tiles (32 KB). Loop-end barrier separates the uses.
    __shared__ __align__(32) unsigned char smemRaw[32768];
    __shared__ float sKp[8][16 * 16];   // per-wave Kp features (8 KB)
    __shared__ float sS[16 * 64];       // block partial S (4 KB)
    __shared__ float sw[64 * 8];        // random-feature matrix (2 KB)

    const int tid  = threadIdx.x;
    const int lane = tid & 31;
    const int wv   = tid >> 5;
    const int lo   = lane & 15;
    const int hi   = lane >> 4;
    const float RS8 = 0.35355339059327373f; // 1/sqrt(8)

    for (int i = tid; i < 512;  i += 256) sw[i] = w[i];
    for (int i = tid; i < 1024; i += 256) sS[i] = 0.0f;

    const int rowbase = blockIdx.x * ROWS_PER_BLK + wv * 16;
    const float* xr = X + (size_t)(rowbase + lo) * NE + (hi << 4);

    v8f acc[12];
    #pragma unroll
    for (int c = 0; c < 12; ++c)
        acc[c] = (v8f){0.f, 0.f, 0.f, 0.f, 0.f, 0.f, 0.f, 0.f};

    unsigned char* sB = smemRaw;
    const unsigned char* wzB = (const unsigned char*)wswz;

#if HAS_ASYNC_LDS
    // prologue: chunk 0 -> buffer 0 (3 x b128 per thread = 12 KB per block)
    #pragma unroll
    for (int j = 0; j < 3; ++j) {
        int off = (tid + j * 256) * 16;
        ASYNC_CP16(wzB + off, sB + off);
    }
#endif

    for (int kc = 0; kc < NKC; ++kc) {
        const int cur = kc & 1;
        unsigned char* bufCur = sB + cur * CHUNK_BYTES;

#if HAS_ASYNC_LDS
        if (kc + 1 < NKC) {
            unsigned char* bufNxt = sB + (cur ^ 1) * CHUNK_BYTES;
            const unsigned char* gsrc = wzB + (size_t)(kc + 1) * CHUNK_BYTES;
            #pragma unroll
            for (int j = 0; j < 3; ++j) {
                int off = (tid + j * 256) * 16;
                ASYNC_CP16(gsrc + off, bufNxt + off);
            }
            WAIT_ASYNC(3);   // current buffer done; keep prefetch in flight
        } else {
            WAIT_ASYNC(0);
        }
        __syncthreads();
#else
        __syncthreads();     // previous reads of bufCur done
        {
            const float4* gsrc = (const float4*)(wzB + (size_t)kc * CHUNK_BYTES);
            float4* ldst = (float4*)bufCur;
            #pragma unroll
            for (int j = 0; j < 3; ++j)
                ldst[tid + j * 256] = gsrc[tid + j * 256];
        }
        __syncthreads();
#endif

        // A fragment: elem e -> K = 16*hi + e
        const float* xp = xr + kc * 32;
        if (kc + 1 < NKC) __builtin_prefetch(xp + 32, 0, 1);  // global_prefetch
        float4 f0 = ((const float4*)xp)[0];
        float4 f1 = ((const float4*)xp)[1];
        float4 f2 = ((const float4*)(xp + 8))[0];
        float4 f3 = ((const float4*)(xp + 8))[1];
        v16bf a;
        a[0]=(__bf16)f0.x;  a[1]=(__bf16)f0.y;  a[2]=(__bf16)f0.z;  a[3]=(__bf16)f0.w;
        a[4]=(__bf16)f1.x;  a[5]=(__bf16)f1.y;  a[6]=(__bf16)f1.z;  a[7]=(__bf16)f1.w;
        a[8]=(__bf16)f2.x;  a[9]=(__bf16)f2.y;  a[10]=(__bf16)f2.z; a[11]=(__bf16)f2.w;
        a[12]=(__bf16)f3.x; a[13]=(__bf16)f3.y; a[14]=(__bf16)f3.z; a[15]=(__bf16)f3.w;

        const v16bf* bfr = (const v16bf*)bufCur;
        #pragma unroll
        for (int ct = 0; ct < 12; ++ct) {
            v16bf bf = bfr[ct * 32 + lane];
            acc[ct] = __builtin_amdgcn_wmma_f32_16x16x32_bf16(
                false, a, false, bf, (short)0, acc[ct], false, false);
        }
        __syncthreads();     // all waves done reading bufCur before it is reused
    }

    // ---------------- epilogue: bias + features + partial S ----------------
    float (*sRow)[16 * 64] = (float (*)[16 * 64])smemRaw;  // reuse arena
    const int fm = lane >> 1;
    const int jb = (lane & 1) * 4;

    // --- q ---
    #pragma unroll
    for (int ct = 0; ct < 4; ++ct) {
        int col = ct * 16 + lo;
        float bb = bq[col];
        #pragma unroll
        for (int i = 0; i < 8; ++i)
            sRow[wv][(i + (hi << 3)) * 64 + col] = acc[ct][i] + bb;
    }
    __syncthreads();
    {
        float p[4] = {0.f, 0.f, 0.f, 0.f};
        #pragma unroll 8
        for (int d = 0; d < 64; ++d) {
            float qv = sRow[wv][fm * 64 + d];
            #pragma unroll
            for (int jj = 0; jj < 4; ++jj)
                p[jj] = fmaf(qv, sw[d * 8 + jb + jj], p[jj]);
        }
        int rg = rowbase + fm;
        #pragma unroll
        for (int jj = 0; jj < 4; ++jj) {
            float sv, cv;
            sincosf(p[jj], &sv, &cv);
            qp[(size_t)rg * 16 + jb + jj]     = cv * RS8;
            qp[(size_t)rg * 16 + jb + jj + 8] = sv * RS8;
        }
    }
    __syncthreads();

    // --- k ---
    #pragma unroll
    for (int ct = 4; ct < 8; ++ct) {
        int col = (ct - 4) * 16 + lo;
        float bb = bk[col];
        #pragma unroll
        for (int i = 0; i < 8; ++i)
            sRow[wv][(i + (hi << 3)) * 64 + col] = acc[ct][i] + bb;
    }
    __syncthreads();
    {
        float p[4] = {0.f, 0.f, 0.f, 0.f};
        #pragma unroll 8
        for (int d = 0; d < 64; ++d) {
            float kv = sRow[wv][fm * 64 + d];
            #pragma unroll
            for (int jj = 0; jj < 4; ++jj)
                p[jj] = fmaf(kv, sw[d * 8 + jb + jj], p[jj]);
        }
        #pragma unroll
        for (int jj = 0; jj < 4; ++jj) {
            float sv, cv;
            sincosf(p[jj], &sv, &cv);
            sKp[wv][fm * 16 + jb + jj]     = cv * RS8;
            sKp[wv][fm * 16 + jb + jj + 8] = sv * RS8;
        }
    }
    __syncthreads();

    // --- v ---
    #pragma unroll
    for (int ct = 8; ct < 12; ++ct) {
        int col = (ct - 8) * 16 + lo;
        float bb = bv[col];
        #pragma unroll
        for (int i = 0; i < 8; ++i)
            sRow[wv][(i + (hi << 3)) * 64 + col] = acc[ct][i] + bb;
    }
    __syncthreads();

    // partial S[j][d] += sum_m Kp[m][j] * v[m][d]; lane owns d = 2*lane, +1
    {
        int d0 = lane << 1;
        float v0[16], v1[16];
        #pragma unroll
        for (int m = 0; m < 16; ++m) {
            v0[m] = sRow[wv][m * 64 + d0];
            v1[m] = sRow[wv][m * 64 + d0 + 1];
        }
        #pragma unroll
        for (int j = 0; j < 16; ++j) {
            float s0 = 0.f, s1 = 0.f;
            #pragma unroll
            for (int m = 0; m < 16; ++m) {
                float kpv = sKp[wv][m * 16 + j];
                s0 = fmaf(kpv, v0[m], s0);
                s1 = fmaf(kpv, v1[m], s1);
            }
            atomicAdd(&sS[j * 64 + d0],     s0);   // ds_add_f32
            atomicAdd(&sS[j * 64 + d0 + 1], s1);
        }
    }
    __syncthreads();

    ((float4*)(sPart + (size_t)blockIdx.x * 1024))[tid] = ((float4*)sS)[tid];
}

// ---------------------------------------------------------------------------
// Kernel 2: reduce 64 block partials per batch -> S[b][16][64]
// ---------------------------------------------------------------------------
__global__ void favor_sreduce_kernel(const float* __restrict__ sPart,
                                     float* __restrict__ S)
{
    int g   = blockIdx.x * blockDim.x + threadIdx.x;   // 0..4095
    int b   = g >> 10;
    int idx = g & 1023;
    float acc = 0.0f;
    #pragma unroll 8
    for (int p = 0; p < 64; ++p)
        acc += sPart[((size_t)(b * 64 + p)) * 1024 + idx];
    S[b * 1024 + idx] = acc;
}

// ---------------------------------------------------------------------------
// Kernel 3: y = qP @ S  (memory-bound; S cached in LDS)
// Block: 256 threads = 4 rows x 64 cols.
// ---------------------------------------------------------------------------
__global__ __launch_bounds__(256) void favor_y_kernel(
    const float* __restrict__ qp,   // [32768][16]
    const float* __restrict__ S,    // [4][16][64]
    float* __restrict__ y)          // [32768][64]
{
    __shared__ float sSf[1024];
    int row0 = blockIdx.x * 4;
    int b    = row0 >> 13;
    ((float4*)sSf)[threadIdx.x] = ((const float4*)(S + (size_t)b * 1024))[threadIdx.x];
    __syncthreads();

    int row = row0 + (threadIdx.x >> 6);
    int d   = threadIdx.x & 63;
    const float* qr = qp + (size_t)row * 16;
    float acc = 0.0f;
    #pragma unroll
    for (int j = 0; j < 16; ++j)
        acc = fmaf(qr[j], sSf[j * 64 + d], acc);
    y[(size_t)row * 64 + d] = acc;
}

// ---------------------------------------------------------------------------
extern "C" void kernel_launch(void* const* d_in, const int* in_sizes, int n_in,
                              void* d_out, int out_size, void* d_ws, size_t ws_size,
                              hipStream_t stream)
{
    (void)in_sizes; (void)n_in; (void)out_size; (void)ws_size;
    const float* x  = (const float*)d_in[0];
    const float* w  = (const float*)d_in[1];
    const float* Wq = (const float*)d_in[2];
    const float* bq = (const float*)d_in[3];
    const float* Wk = (const float*)d_in[4];
    const float* bk = (const float*)d_in[5];
    const float* Wv = (const float*)d_in[6];
    const float* bv = (const float*)d_in[7];

    float*  qp    = (float*)d_ws;              // 32768*16 = 524288 f
    float*  sPart = qp + 32768 * 16;           // 256*1024 = 262144 f
    float*  Sf    = sPart + 256 * 1024;        // 4*1024   =   4096 f
    __bf16* wswz  = (__bf16*)(Sf + 4096);      // 196608 bf16 = 384 KB (16B aligned)
    float*  y     = (float*)d_out;

    favor_wswz_kernel<<<48, 256, 0, stream>>>(Wq, Wk, Wv, wswz);
    favor_proj_kernel<<<256, 256, 0, stream>>>(x, w, bq, bk, bv, wswz, qp, sPart);
    favor_sreduce_kernel<<<16, 256, 0, stream>>>(sPart, Sf);
    favor_y_kernel<<<8192, 256, 0, stream>>>(qp, Sf, y);
}
